// AttentionalFactorizationMachineLayer_86844238725596
// MI455X (gfx1250) — matrix operand order repaired and checked by reading, not compiled, not run on hardware
//
#include <hip/hip_runtime.h>
#include <hip/hip_bf16.h>

#define BATCH  1024
#define NF     64
#define EMB    64
#define ATT    64
#define NPAIR  2016
#define NTILES 126      // 2016 / 16
#define TPB    256
#define WAVES  8

typedef __attribute__((ext_vector_type(16))) __bf16 v16bf;
typedef __attribute__((ext_vector_type(8)))  float  v8f;
typedef __attribute__((ext_vector_type(8)))  unsigned v8u;

// Pack two fp32 into one 32-bit word of two bf16 (native cvt; backend may
// fuse adjacent converts into v_cvt_pk_bf16_f32).
__device__ __forceinline__ unsigned pk2(float a, float b) {
  unsigned short lo = __builtin_bit_cast(unsigned short, (__bf16)a);
  unsigned short hi = __builtin_bit_cast(unsigned short, (__bf16)b);
  return (unsigned)lo | ((unsigned)hi << 16);
}

// Fill 4 consecutive packed words (8 bf16 elements) of an A-fragment with
// products of two LDS rows, starting at float4 block index b.
template<int WOFF>
__device__ __forceinline__ void prod8(const float4* ei, const float4* ej,
                                      int b, v8u& W) {
  float4 x0 = ei[b], x1 = ei[b + 1];
  float4 y0 = ej[b], y1 = ej[b + 1];
  W[WOFF + 0] = pk2(x0.x * y0.x, x0.y * y0.y);
  W[WOFF + 1] = pk2(x0.z * y0.z, x0.w * y0.w);
  W[WOFF + 2] = pk2(x1.x * y1.x, x1.y * y1.y);
  W[WOFF + 3] = pk2(x1.z * y1.z, x1.w * y1.w);
}

// Build triu(k=1) pair table once: ptab[p] = (i<<8)|j
__global__ void afm_pair_table(unsigned* __restrict__ ptab) {
  int p = blockIdx.x * blockDim.x + threadIdx.x;
  if (p >= NPAIR) return;
  int q = NPAIR - 1 - p;                       // count from the end
  int k = (int)((__fsqrt_rn(8.0f * (float)q + 1.0f) - 1.0f) * 0.5f);
  while (k * (k + 1) / 2 > q) --k;             // fix float rounding
  while ((k + 1) * (k + 2) / 2 <= q) ++k;
  int i = NF - 2 - k;
  int r = q - k * (k + 1) / 2;
  int j = NF - 1 - r;
  ptab[p] = ((unsigned)i << 8) | (unsigned)j;
}

__global__ __launch_bounds__(TPB) void afm_kernel(
    const float* __restrict__ emb, const float* __restrict__ W1,
    const float* __restrict__ b1,  const float* __restrict__ w2,
    const float* __restrict__ b2,  const unsigned* __restrict__ ptab,
    float* __restrict__ out, float* __restrict__ attn_out) {
  __shared__ float          sEmb[NF * EMB];        // 16 KB, fp32 batch row
  __shared__ unsigned short sW1t[ATT * EMB];       // 8 KB, W1^T in bf16 [a][e]
  __shared__ float          sLg[NPAIR];            // logits -> attn (in place)
  __shared__ unsigned       sPair[NPAIR];          // (i<<8)|j
  __shared__ float          sB1[ATT];
  __shared__ float          sW2[ATT];
  __shared__ float          sRed[TPB];
  __shared__ float          sPart[WAVES * EMB];    // deterministic reduction

  const int tid   = threadIdx.x;
  const int lane  = tid & 31;
  const int wave  = tid >> 5;
  const int col16 = lane & 15;
  const int half  = lane >> 4;
  const int bidx  = blockIdx.x;

  // ---- stage inputs into LDS ----
  const float4* eg4 = (const float4*)(emb + (size_t)bidx * (NF * EMB));
  float4* es4 = (float4*)sEmb;
  for (int v = tid; v < NF * EMB / 4; v += TPB) es4[v] = eg4[v];
  for (int idx = tid; idx < EMB * ATT; idx += TPB) {
    int e = idx >> 6, a = idx & 63;                // W1[e][a] row-major
    sW1t[a * EMB + e] =
        __builtin_bit_cast(unsigned short, (__bf16)W1[idx]);  // transposed
  }
  for (int p = tid; p < NPAIR; p += TPB) sPair[p] = ptab[p];
  if (tid < ATT) { sB1[tid] = b1[tid]; sW2[tid] = w2[tid]; }
  const float b2v = b2[0];
  __syncthreads();

  // ---- B fragments (W1) in registers, B 32x16 bf16, contiguous K order (matches A):
  //      lane half selects K block of 16; element e -> K = kh*32 + half*16 + e
  v16bf Bf[4][2];
#pragma unroll
  for (int nt = 0; nt < 4; ++nt)
#pragma unroll
    for (int kh = 0; kh < 2; ++kh)
      Bf[nt][kh] =
          *(const v16bf*)&sW1t[(nt * 16 + col16) * EMB + kh * 32 + half * 16];

  // ---- phase 1: logits via bf16 WMMA over 16-pair tiles ----
  for (int t = wave; t < NTILES; t += WAVES) {
    unsigned pr = sPair[t * 16 + col16];           // A row M = lane&15
    const float4* ei = (const float4*)&sEmb[(pr >> 8) * EMB];
    const float4* ej = (const float4*)&sEmb[(pr & 255u) * EMB];
    // A 16x32 bf16 layout (contiguous, matches B): e -> K=half*16+e
    v8u W0, W1f;
    prod8<0>(ei, ej, half * 4, W0);                // K 0..7  | 16..23
    prod8<4>(ei, ej, 2 + half * 4, W0);            // K 8..15 | 24..31
    prod8<0>(ei, ej, 8 + half * 4, W1f);           // K 32..39| 48..55
    prod8<4>(ei, ej, 10 + half * 4, W1f);          // K 40..47| 56..63
    v16bf A0 = __builtin_bit_cast(v16bf, W0);
    v16bf A1 = __builtin_bit_cast(v16bf, W1f);

    float acc[8];
#pragma unroll
    for (int r = 0; r < 8; ++r) acc[r] = 0.f;

#pragma unroll
    for (int nt = 0; nt < 4; ++nt) {
      v8f D = {};
      D = __builtin_amdgcn_wmma_f32_16x16x32_bf16(false, A0, false, Bf[nt][0],
                                                  (short)0, D, false, false);
      D = __builtin_amdgcn_wmma_f32_16x16x32_bf16(false, A1, false, Bf[nt][1],
                                                  (short)0, D, false, false);
      int col = nt * 16 + col16;                   // D col N = lane&15
      float bb = sB1[col], ww = sW2[col];
#pragma unroll
      for (int r = 0; r < 8; ++r) {                // D row M = half*8 + r
        float h = D[r] + bb;
        h = h > 0.f ? h : 0.f;
        acc[r] += h * ww;
      }
    }
    // reduce over N (16 lanes within each half of the wave)
#pragma unroll
    for (int r = 0; r < 8; ++r) {
      acc[r] += __shfl_xor(acc[r], 1, 32);
      acc[r] += __shfl_xor(acc[r], 2, 32);
      acc[r] += __shfl_xor(acc[r], 4, 32);
      acc[r] += __shfl_xor(acc[r], 8, 32);
    }
    if (col16 == 0) {
#pragma unroll
      for (int r = 0; r < 8; ++r)
        sLg[t * 16 + half * 8 + r] = acc[r] + b2v;
    }
  }
  __syncthreads();

  // ---- phase 2: softmax over 2016 logits (deterministic tree) ----
  float lmax = -__builtin_inff();
  for (int p = tid; p < NPAIR; p += TPB) lmax = fmaxf(lmax, sLg[p]);
  sRed[tid] = lmax;
  __syncthreads();
  for (int s = TPB / 2; s > 0; s >>= 1) {
    if (tid < s) sRed[tid] = fmaxf(sRed[tid], sRed[tid + s]);
    __syncthreads();
  }
  const float gmax = sRed[0];
  __syncthreads();

  float lsum = 0.f;
  for (int p = tid; p < NPAIR; p += TPB) {
    float e = __expf(sLg[p] - gmax);
    sLg[p] = e;
    lsum += e;
  }
  sRed[tid] = lsum;
  __syncthreads();
  for (int s = TPB / 2; s > 0; s >>= 1) {
    if (tid < s) sRed[tid] += sRed[tid + s];
    __syncthreads();
  }
  const float inv = 1.0f / sRed[0];
  float* attnB = attn_out + (size_t)bidx * NPAIR;
  for (int p = tid; p < NPAIR; p += TPB) {
    float a = sLg[p] * inv;
    sLg[p] = a;
    attnB[p] = a;                                  // second output tensor
  }
  __syncthreads();

  // ---- phase 3: outputs[b,:] = sum_p attn[p] * inner[p,:] ----
  // lane owns elements e0 = 2*lane, e0+1 -> one ds_load_b64 per row
  const int e0 = 2 * lane;
  float a0 = 0.f, a1 = 0.f;
  for (int p = wave; p < NPAIR; p += WAVES) {
    unsigned pr = sPair[p];
    int i = pr >> 8, j = pr & 255u;
    float w = sLg[p];
    float2 xi = *(const float2*)&sEmb[i * EMB + e0];
    float2 xj = *(const float2*)&sEmb[j * EMB + e0];
    a0 += w * xi.x * xj.x;
    a1 += w * xi.y * xj.y;
  }
  sPart[wave * EMB + e0]     = a0;
  sPart[wave * EMB + e0 + 1] = a1;
  __syncthreads();
  if (tid < EMB) {
    float s = 0.f;
#pragma unroll
    for (int w = 0; w < WAVES; ++w) s += sPart[w * EMB + tid];
    out[(size_t)bidx * EMB + tid] = s;
  }
}

extern "C" void kernel_launch(void* const* d_in, const int* in_sizes, int n_in,
                              void* d_out, int out_size, void* d_ws,
                              size_t ws_size, hipStream_t stream) {
  const float* emb = (const float*)d_in[0];  // (1024, 64, 64)
  const float* W1  = (const float*)d_in[1];  // (64, 64)
  const float* b1  = (const float*)d_in[2];  // (64,)
  const float* w2  = (const float*)d_in[3];  // (64,)
  const float* b2  = (const float*)d_in[4];  // scalar
  float* out = (float*)d_out;                // outputs (1024*64) then attn
  float* attn = out + (size_t)BATCH * EMB;   // (1024*2016)
  unsigned* ptab = (unsigned*)d_ws;          // 2016 * 4 bytes

  afm_pair_table<<<(NPAIR + 255) / 256, 256, 0, stream>>>(ptab);
  afm_kernel<<<BATCH, TPB, 0, stream>>>(emb, W1, b1, w2, b2, ptab, out, attn);
}
